// FlashAttention_49228915147338
// MI455X (gfx1250) — matrix operand order repaired and checked by reading, not compiled, not run on hardware
//
#include <hip/hip_runtime.h>
#include <hip/hip_bf16.h>

typedef __attribute__((ext_vector_type(16))) _Float16 v16h;
typedef __attribute__((ext_vector_type(8)))  _Float16 v8h;
typedef __attribute__((ext_vector_type(4)))  _Float16 v4h;
typedef __attribute__((ext_vector_type(8)))  float    v8f;
typedef __attribute__((ext_vector_type(4)))  float    v4f;

#define SQ   4096
#define SK   4096
#define DH   128
#define BM   128    // query rows per workgroup (8 waves x 16 rows)
#define BN   64     // KV tile (must match reference BLOCK_N for numerics)
#define NTHR 256    // 8 wave32

union FragA { v16h v; v8h h[2]; };

__global__ __launch_bounds__(NTHR)
void FlashAttention_49228915147338_kernel(const float* __restrict__ Q,
                                          const float* __restrict__ K,
                                          const float* __restrict__ V,
                                          float* __restrict__ O)
{
    // smem0 first holds Q*scale (f16, 32KB); after Q fragments are in registers
    // it is reused as the per-wave P bounce buffers (8 x 16 x BN = 16KB).
    __shared__ __align__(32) _Float16 smem0[BM * DH];     // 32 KB
    __shared__ __align__(32) _Float16 ks[BN * DH];        // 16 KB
    __shared__ __align__(32) _Float16 vts[DH * BN];       // 16 KB, vts[d][kv]

    const int tid  = threadIdx.x;
    const int lane = tid & 31;
    const int wave = tid >> 5;
    const int bh   = blockIdx.y;      // batch*head
    const int qt   = blockIdx.x;      // query tile

    const float scale = 0.08838834764831845f;   // 1/sqrt(128), folded into Q

    const size_t qbase  = ((size_t)bh * SQ + (size_t)qt * BM) * DH;
    const size_t kvbase = (size_t)bh * SK * DH;

    // ---- stage Q tile (f32 -> f16, scale folded in), vectorized ----
    for (int j = tid; j < BM * DH / 4; j += NTHR) {
        const v4f f = *(const v4f*)&Q[qbase + (size_t)j * 4];
        v4h h;
        h[0] = (_Float16)(f[0] * scale);
        h[1] = (_Float16)(f[1] * scale);
        h[2] = (_Float16)(f[2] * scale);
        h[3] = (_Float16)(f[3] * scale);
        *(v4h*)&smem0[j * 4] = h;
    }
    __syncthreads();

    // ---- load Q A-fragments for this wave's 16 rows ----
    const int mrow = lane & 15;          // row within 16-tile (A/B col select)
    const int koff = (lane >> 4) * 8;    // A-layout k offset
    const int rsel = lane >> 4;          // C-layout row select (rows i / i+8)

    FragA qa[4];
#pragma unroll
    for (int c = 0; c < 4; ++c) {
        const int b = (wave * 16 + mrow) * DH + c * 32 + koff;
        qa[c].h[0] = *(const v8h*)&smem0[b];
        qa[c].h[1] = *(const v8h*)&smem0[b + 16];
    }
    // per-wave P bounce region (aliases qs; ordered by the in-loop barrier)
    _Float16* const pw = smem0 + wave * (16 * BN);

    const v8f vzero = {};
    v8f acc[8];
#pragma unroll
    for (int d = 0; d < 8; ++d) acc[d] = vzero;
    float m_prev[8], l_prev[8];
#pragma unroll
    for (int i = 0; i < 8; ++i) { m_prev[i] = -__builtin_inff(); l_prev[i] = 0.0f; }

    for (int kb = 0; kb < SK / BN; ++kb) {
        const size_t g = kvbase + (size_t)kb * BN * DH;

        // ---- stage K tile (vectorized f32x4 -> f16x4): 8 chunks/thread ----
        for (int j = tid; j < BN * DH / 4; j += NTHR) {
            const v4f f = *(const v4f*)&K[g + (size_t)j * 4];
            v4h h;
            h[0] = (_Float16)f[0];
            h[1] = (_Float16)f[1];
            h[2] = (_Float16)f[2];
            h[3] = (_Float16)f[3];
            *(v4h*)&ks[j * 4] = h;
        }
        // ---- stage V transposed via 4x4 register blocks: 2 blocks/thread ----
        // 512 blocks: kv-block = b/32, d-block = b%32 (lanes sweep d -> coalesced)
        for (int b = tid; b < (BN / 4) * (DH / 4); b += NTHR) {
            const int kv0 = (b >> 5) << 2;
            const int d0  = (b & 31) << 2;
            const v4f r0 = *(const v4f*)&V[g + (size_t)(kv0 + 0) * DH + d0];
            const v4f r1 = *(const v4f*)&V[g + (size_t)(kv0 + 1) * DH + d0];
            const v4f r2 = *(const v4f*)&V[g + (size_t)(kv0 + 2) * DH + d0];
            const v4f r3 = *(const v4f*)&V[g + (size_t)(kv0 + 3) * DH + d0];
#pragma unroll
            for (int dd = 0; dd < 4; ++dd) {
                v4h h;
                h[0] = (_Float16)r0[dd];
                h[1] = (_Float16)r1[dd];
                h[2] = (_Float16)r2[dd];
                h[3] = (_Float16)r3[dd];
                *(v4h*)&vts[(d0 + dd) * BN + kv0] = h;
            }
        }
        __syncthreads();

        // ---- S = (Q*scale) @ K^T : 4 col-tiles x 4 k-chunks of WMMA ----
        v8f Ct[4];
#pragma unroll
        for (int nt = 0; nt < 4; ++nt) {
            v8f c = vzero;
#pragma unroll
            for (int ch = 0; ch < 4; ++ch) {
                FragA kf;   // B = K^T: column n = kv row, contiguous over d
                const int b = (nt * 16 + mrow) * DH + ch * 32 + (lane >> 4) * 8;
                kf.h[0] = *(const v8h*)&ks[b];
                kf.h[1] = *(const v8h*)&ks[b + 16];
                c = __builtin_amdgcn_wmma_f32_16x16x32_f16(
                        false, qa[ch].v, false, kf.v, (short)0, c, false, false);
            }
            Ct[nt] = c;
        }

        // ---- online softmax (reference's block-wise variant) ----
#pragma unroll
        for (int i = 0; i < 8; ++i) {
            float r = fmaxf(fmaxf(Ct[0][i], Ct[1][i]), fmaxf(Ct[2][i], Ct[3][i]));
#pragma unroll
            for (int m = 8; m >= 1; m >>= 1)
                r = fmaxf(r, __shfl_xor(r, m, 32));
            const float mc = fmaxf(r, m_prev[i]);

            float p[4];
            float ls = 0.0f;
#pragma unroll
            for (int nt = 0; nt < 4; ++nt) { p[nt] = __expf(Ct[nt][i] - mc); ls += p[nt]; }
#pragma unroll
            for (int m = 8; m >= 1; m >>= 1)
                ls += __shfl_xor(ls, m, 32);

            const float alpha = __expf(m_prev[i] - mc);
            const float ln = l_prev[i] * alpha + ls;
            m_prev[i] = mc;
            l_prev[i] = ln;
            const float inv = 1.0f / ln;

            const int row = i + 8 * rsel;
#pragma unroll
            for (int nt = 0; nt < 4; ++nt)
                pw[row * BN + nt * 16 + mrow] = (_Float16)(p[nt] * inv);
        }
        __syncthreads();   // P visible across lanes

        // ---- acc += P @ V : 8 dout-tiles x 2 k-chunks of WMMA ----
        FragA pa[2];
#pragma unroll
        for (int ch = 0; ch < 2; ++ch) {
            const int b = mrow * BN + ch * 32 + koff;
            pa[ch].h[0] = *(const v8h*)&pw[b];
            pa[ch].h[1] = *(const v8h*)&pw[b + 16];
        }
#pragma unroll
        for (int dt = 0; dt < 8; ++dt) {
#pragma unroll
            for (int ch = 0; ch < 2; ++ch) {
                FragA vf;   // B = V: column n = dout, contiguous over kv in vts
                const int b = (dt * 16 + mrow) * BN + ch * 32 + (lane >> 4) * 8;
                vf.h[0] = *(const v8h*)&vts[b];
                vf.h[1] = *(const v8h*)&vts[b + 16];
                acc[dt] = __builtin_amdgcn_wmma_f32_16x16x32_f16(
                            false, pa[ch].v, false, vf.v, (short)0, acc[dt], false, false);
            }
        }
        __syncthreads();   // done reading ks/vts/pw before next staging
    }

    // ---- write out (fp32, streaming/non-temporal) ----
    const size_t obase = ((size_t)bh * SQ + (size_t)qt * BM + wave * 16) * DH;
#pragma unroll
    for (int dt = 0; dt < 8; ++dt)
#pragma unroll
        for (int i = 0; i < 8; ++i)
            __builtin_nontemporal_store(
                acc[dt][i],
                &O[obase + (size_t)(i + 8 * rsel) * DH + dt * 16 + mrow]);
}

extern "C" void kernel_launch(void* const* d_in, const int* in_sizes, int n_in,
                              void* d_out, int out_size, void* d_ws, size_t ws_size,
                              hipStream_t stream) {
    const float* q = (const float*)d_in[0];
    const float* k = (const float*)d_in[1];
    const float* v = (const float*)d_in[2];
    float* o = (float*)d_out;

    dim3 grid(SQ / BM, 2 * 16);   // (query tiles, B*H)
    dim3 block(NTHR);
    FlashAttention_49228915147338_kernel<<<grid, block, 0, stream>>>(q, k, v, o);
}
